// rits_18897856102806
// MI455X (gfx1250) — compile-verified, matrix operand order checked by reading, not run on hardware
//
#include <hip/hip_runtime.h>
#include <hip/hip_bf16.h>
#include <cstdint>
#include <cstddef>

// Problem dims
#define B_N 2048
#define T_N 48
#define V_N 59
#define H_N 512
#define MTILE 16
#define THREADS 256

typedef __attribute__((ext_vector_type(16))) __bf16 v16bf;
typedef __attribute__((ext_vector_type(8)))  __bf16 v8bf;
typedef __attribute__((ext_vector_type(8)))  float  v8f;
typedef __attribute__((ext_vector_type(4)))  unsigned int u32x4;
typedef __attribute__((ext_vector_type(8)))  int i32x8;
typedef __attribute__((ext_vector_type(4)))  int i32x4;

// ---------------- workspace layout (bytes) ----------------
// packed bf16 weight fragments: tile = 32 lanes x 16 halves = 512 halves
#define OFF_MSUM  0                                   // 48 f32
#define OFF_WGH   256                                 // 2 Kt * 32 Nt
#define OFF_WHIST (OFF_WGH   + 2*32*512*2)            // 16 Kt * 4 Nt
#define OFF_WFR   (OFF_WHIST + 16*4*512*2)            // 2 Kt * 4 Nt
#define OFF_WWC   (OFF_WFR   + 2*4*512*2)             // 4 Kt * 4 Nt
#define OFF_WCAT  (OFF_WWC   + 4*4*512*2)             // 20 Kt * 128 Nt
#define WS_NEED   (OFF_WCAT  + 20*128*512*2)          // ~2.78 MB

// ---------------- LDS layout (bytes) inside rits_main ----------------
#define L_HS    0                       // [16][512] f32
#define L_CS    32768                   // [16][512] f32
#define L_XH    65536                   // [16][64]  f32
#define L_XU    69632                   // [16][64]  f32
#define L_BT    73728                   // [16][64]  f32
#define L_RED   77824                   // [256]     f32
#define L_INVM  78848                   // [1] f32 (pad 32)
#define L_BSUM  78880                   // [2048] f32 (b_ih + b_hh)
#define L_A5    87072                   // [16][640] bf16 = [x_imp|pad|m|pad|h]
#define L_DB    107552                  // [16][64]  bf16
#define L_ARB   109600                  // [16][64]  bf16
#define L_AWB   111648                  // [16][128] bf16
#define L_STG   115744                  // 2 bufs x 3 arrays x 960 f32 (x,m,d tiles)
#define STG_STRIDE 3840                 // 960 f32
#define SMEM_BYTES (L_STG + 2*3*STG_STRIDE)   // 138784

// ---------------- device helpers ----------------
__device__ inline v8f wmma_bf(v16bf a, v16bf b, v8f c) {
  // emits v_wmma_f32_16x16x32_bf16
  return __builtin_amdgcn_wmma_f32_16x16x32_bf16(
      /*neg_a=*/false, a, /*neg_b=*/false, b,
      /*c_mod=*/(short)0, c, /*reuse_a=*/false, /*reuse_b=*/false);
}

// A-fragment (16x32 bf16) from LDS. Lanes 0-15: rows m=lane, K halves
// kb..kb+7 and kb+16..kb+23 with kb = (lane>=16)?8:0.
__device__ inline v16bf load_afrag(const __bf16* A, int stride, int lane, int colbase) {
  int m  = lane & 15;
  int kb = (lane & 16) ? 8 : 0;
  const __bf16* p = A + m * stride + colbase + kb;
  v8bf lo = *(const v8bf*)p;
  v8bf hi = *(const v8bf*)(p + 16);
  v16bf r;
#pragma unroll
  for (int e = 0; e < 8; ++e) { r[e] = lo[e]; r[8 + e] = hi[e]; }
  return r;
}

// B-fragment (32x16 bf16), pre-packed fragment-major in global (L2-resident):
// 16 contiguous halves per lane => two global_load_b128.
__device__ inline v16bf load_bfrag(const __bf16* W, int tile, int lane) {
  return *(const v16bf*)(W + (size_t)tile * 512 + lane * 16);
}

__device__ inline float sigm(float v) { return 1.f / (1.f + __expf(-v)); }

// TDM: DMA a 2D f32 tile (rows x rowlen, row stride in elements) from global
// into LDS at byte offset lds_off. Descriptor per CDNA5 ISA §8 (D# groups).
__device__ inline void tdm_load_2d(const void* gptr, unsigned lds_off,
                                   unsigned rowlen, unsigned rows,
                                   unsigned row_stride) {
  unsigned long long ga = (unsigned long long)(uintptr_t)gptr;
  u32x4 g0;
  g0[0] = 1u;                                         // count=1, user mode
  g0[1] = lds_off;                                    // lds_addr (bytes)
  g0[2] = (unsigned)(ga & 0xFFFFFFFFull);             // global_addr[31:0]
  g0[3] = (unsigned)((ga >> 32) & 0x1FFFFFFull)       // global_addr[56:32]
        | (2u << 30);                                 // type = 2 ("image")
  i32x8 g1;
  g1[0] = (int)(2u << 16);                            // data_size = 4 bytes
  g1[1] = (int)((rowlen & 0xFFFFu) << 16);            // tensor_dim0[15:0]
  g1[2] = (int)(((rowlen >> 16) & 0xFFFFu) | ((rows & 0xFFFFu) << 16)); // dim0 hi | dim1 lo
  g1[3] = (int)(((rows >> 16) & 0xFFFFu) | ((rowlen & 0xFFFFu) << 16)); // dim1 hi | tile_dim0
  g1[4] = (int)(rows & 0xFFFFu);                      // tile_dim1 (tile_dim2=0)
  g1[5] = (int)row_stride;                            // tensor_dim0_stride[31:0]
  g1[6] = 0;                                          // stride hi | dim1_stride lo
  g1[7] = 0;
  i32x4 z4 = {0, 0, 0, 0};
#if __clang_major__ >= 23
  i32x8 z8 = {0, 0, 0, 0, 0, 0, 0, 0};
  __builtin_amdgcn_tensor_load_to_lds(g0, g1, z4, z4, z8, 0);
#else
  __builtin_amdgcn_tensor_load_to_lds(g0, g1, z4, z4, 0);
#endif
}

// ---------------- init: zero msum + loss slot ----------------
__global__ void rits_init_kernel(float* msum, float* out) {
  int i = threadIdx.x;
  if (i < T_N) msum[i] = 0.f;
  if (i == 63) out[(size_t)B_N * T_N * V_N] = 0.f;
}

// ---------------- per-timestep mask sums ----------------
__global__ void rits_msum_kernel(const float* __restrict__ mask, float* __restrict__ msum) {
  __shared__ float sred[THREADS];
  int t = blockIdx.x;
  int b = blockIdx.y * THREADS + threadIdx.x;
  float s = 0.f;
  const float* p = mask + (size_t)b * T_N * V_N + (size_t)t * V_N;
  for (int v = 0; v < V_N; ++v) s += p[v];
  sred[threadIdx.x] = s;
  __syncthreads();
  for (int st = THREADS / 2; st > 0; st >>= 1) {
    if (threadIdx.x < st) sred[threadIdx.x] += sred[threadIdx.x + st];
    __syncthreads();
  }
  if (threadIdx.x == 0) atomicAdd(&msum[t], sred[0]);
}

// ---------------- weight pack: f32 -> bf16 fragment-major B panels ----------
// dst element gid = ((kt*Ntiles + nt)*32 + lane)*16 + e
__global__ void rits_pack_kernel(__bf16* __restrict__ dst,
                                 const float* __restrict__ S1,
                                 const float* __restrict__ S2,
                                 int mode, int Ktiles, int Ntiles) {
  size_t total = (size_t)Ktiles * Ntiles * 512;
  size_t gid = (size_t)blockIdx.x * THREADS + threadIdx.x;
  if (gid >= total) return;
  int e    = (int)(gid & 15);
  int lane = (int)((gid >> 4) & 31);
  size_t tile = gid >> 9;
  int nt = (int)(tile % Ntiles);
  int kt = (int)(tile / Ntiles);
  int kb   = (lane & 16) ? 8 : 0;
  int kidx = (e < 8) ? (kb + e) : (kb + 16 + (e - 8));
  int k = kt * 32 + kidx;
  int n = nt * 16 + (lane & 15);
  float v = 0.f;
  switch (mode) {
    case 0:  // W_gh [512][59] -> panel K=64(pad), N=512
      if (k < V_N) v = S1[n * V_N + k];
      break;
    case 1:  // W_hist [59][512] -> K=512, N=64(pad)
      if (n < V_N) v = S1[n * H_N + k];
      break;
    case 2:  // W_fr [59][59], zero diagonal -> K=64, N=64
      if (k < V_N && n < V_N && k != n) v = S1[n * V_N + k];
      break;
    case 3:  // W_wc [59][118]; A = [gamma_x(0..58) pad | m(64..122) pad]
      if (n < V_N) {
        if (k < V_N)                 v = S1[n * 118 + k];
        else if (k >= 64 && k < 123) v = S1[n * 118 + (k - 5)];
      }
      break;
    case 4:  // fused [W_ih | W_hh]: A = [x_imp|pad|m|pad|h], K=640, N=2048
      if (k < V_N)                 v = S1[n * 118 + k];
      else if (k >= 64 && k < 123) v = S1[n * 118 + (k - 5)];
      else if (k >= 128)           v = S2[n * H_N + (k - 128)];
      break;
  }
  dst[gid] = (__bf16)v;
}

// ---------------- main persistent recurrent kernel ----------------
// 128 blocks x 256 threads (8 waves); block owns 16 batch rows for all steps.
// x/mask/deltas tiles are TDM-prefetched one step ahead into double-buffered LDS.
__global__ __launch_bounds__(THREADS) void rits_main_kernel(
    const float* __restrict__ x, const float* __restrict__ mask, const float* __restrict__ deltas,
    const float* __restrict__ b_gh, const float* __restrict__ w_gx, const float* __restrict__ b_gx,
    const float* __restrict__ b_hist, const float* __restrict__ b_fr, const float* __restrict__ b_wc,
    const float* __restrict__ b_ih, const float* __restrict__ b_hh,
    const __bf16* __restrict__ Wgh, const __bf16* __restrict__ Whist, const __bf16* __restrict__ Wfr,
    const __bf16* __restrict__ Wwc, const __bf16* __restrict__ Wcat,
    const float* __restrict__ msum, float* __restrict__ out) {
  extern __shared__ char smem[];
  float*  hS    = (float*)(smem + L_HS);
  float*  cS    = (float*)(smem + L_CS);
  float*  xhS   = (float*)(smem + L_XH);
  float*  xuS   = (float*)(smem + L_XU);
  float*  btS   = (float*)(smem + L_BT);
  float*  red   = (float*)(smem + L_RED);
  float*  invm  = (float*)(smem + L_INVM);
  float*  bsumS = (float*)(smem + L_BSUM);
  __bf16* A5    = (__bf16*)(smem + L_A5);
  __bf16* dB    = (__bf16*)(smem + L_DB);
  __bf16* arB   = (__bf16*)(smem + L_ARB);
  __bf16* awB   = (__bf16*)(smem + L_AWB);

  const int tid  = threadIdx.x;
  const int lane = tid & 31;
  const int w    = tid >> 5;
  const int r0   = blockIdx.x * MTILE;
  const size_t XIMP_N = (size_t)B_N * T_N * V_N;
  float* outH = out + XIMP_N + 1;
  const size_t gbase = (size_t)r0 * T_N * V_N;

  // kick off TDM prefetch of the t=0 tiles (wave 0 only; EXEC ignored by TDM)
  if (tid < 32) {
    tdm_load_2d(x + gbase,      L_STG + 0 * STG_STRIDE, V_N, MTILE, T_N * V_N);
    tdm_load_2d(mask + gbase,   L_STG + 1 * STG_STRIDE, V_N, MTILE, T_N * V_N);
    tdm_load_2d(deltas + gbase, L_STG + 2 * STG_STRIDE, V_N, MTILE, T_N * V_N);
  }

  // zero state + padded bf16 staging (pads stay zero forever); fused LSTM bias
  for (int i = tid; i < MTILE * H_N; i += THREADS) { hS[i] = 0.f; cS[i] = 0.f; }
  for (int i = tid; i < MTILE * 640; i += THREADS) A5[i] = (__bf16)0.f;
  for (int i = tid; i < MTILE * 64;  i += THREADS) { dB[i] = (__bf16)0.f; arB[i] = (__bf16)0.f; }
  for (int i = tid; i < MTILE * 128; i += THREADS) awB[i] = (__bf16)0.f;
  for (int i = tid; i < 4 * H_N;     i += THREADS) bsumS[i] = b_ih[i] + b_hh[i];
  float lossAcc = 0.f;
  __syncthreads();

  for (int t = 0; t < T_N; ++t) {
    const int cur = t & 1;
    const float* xg = (const float*)(smem + L_STG + (cur * 3 + 0) * STG_STRIDE);
    const float* mg = (const float*)(smem + L_STG + (cur * 3 + 1) * STG_STRIDE);
    const float* dg = (const float*)(smem + L_STG + (cur * 3 + 2) * STG_STRIDE);

    // ---- wait for this step's TDM tiles, then prefetch next step ----
    if (tid < 32) __builtin_amdgcn_s_wait_tensorcnt((short)0);
    __syncthreads();
    if (tid < 32 && (t + 1) < T_N) {
      unsigned nb = L_STG + ((t + 1) & 1) * 3 * STG_STRIDE;
      size_t g = gbase + (size_t)(t + 1) * V_N;
      tdm_load_2d(x + g,      nb + 0 * STG_STRIDE, V_N, MTILE, T_N * V_N);
      tdm_load_2d(mask + g,   nb + 1 * STG_STRIDE, V_N, MTILE, T_N * V_N);
      tdm_load_2d(deltas + g, nb + 2 * STG_STRIDE, V_N, MTILE, T_N * V_N);
    }

    // ---- phase 1: bf16 deltas operand + loss scale ----
    if (tid == 0) invm[0] = 1.f / (msum[t] + 1e-5f);
    for (int i = tid; i < MTILE * V_N; i += THREADS) {
      int r = i / V_N, v = i - r * V_N;
      dB[r * 64 + v] = (__bf16)dg[i];
    }
    __syncthreads();

    // ---- G1: gamma_h = exp(-relu(d@Wgh^T+b)); h *= gamma_h; refresh A5[h] ----
    {
      v16bf a0 = load_afrag(dB, 64, lane, 0);
      v16bf a1 = load_afrag(dB, 64, lane, 32);
#pragma unroll
      for (int j = 0; j < 4; ++j) {
        int nt = w * 4 + j;
        v8f acc = {};
        acc = wmma_bf(a0, load_bfrag(Wgh, 0 * 32 + nt, lane), acc);
        acc = wmma_bf(a1, load_bfrag(Wgh, 1 * 32 + nt, lane), acc);
        int n = nt * 16 + (lane & 15);
        int mrow0 = (lane & 16) ? 8 : 0;
        float bg = b_gh[n];
#pragma unroll
        for (int i = 0; i < 8; ++i) {
          int mr = mrow0 + i;
          float gma = __expf(-fmaxf(acc[i] + bg, 0.f));
          float hv = hS[mr * H_N + n] * gma;
          hS[mr * H_N + n] = hv;
          A5[mr * 640 + 128 + n] = (__bf16)hv;
        }
      }
    }
    __syncthreads();

    // ---- G2: x_h = h @ Whist^T + b_hist (waves 0..3, K=512) ----
    if (w < 4) {
      int nt = w;
      v8f acc = {};
#pragma unroll
      for (int kt = 0; kt < 16; ++kt) {
        v16bf a = load_afrag(A5, 640, lane, 128 + kt * 32);
        acc = wmma_bf(a, load_bfrag(Whist, kt * 4 + nt, lane), acc);
      }
      int n = nt * 16 + (lane & 15);
      int mrow0 = (lane & 16) ? 8 : 0;
      float bh = (n < V_N) ? b_hist[n] : 0.f;
#pragma unroll
      for (int i = 0; i < 8; ++i)
        xhS[(mrow0 + i) * 64 + n] = (n < V_N) ? (acc[i] + bh) : 0.f;
    }
    __syncthreads();

    // ---- phase A: x_r, gamma_x -> bf16 A operands ----
    for (int i = tid; i < MTILE * V_N; i += THREADS) {
      int r = i / V_N, v = i - r * V_N;
      float mv = mg[i];
      float xr = mv * xg[i] + (1.f - mv) * xhS[r * 64 + v];
      arB[r * 64 + v] = (__bf16)xr;
      float gx = __expf(-fmaxf(dg[i] * w_gx[v] + b_gx[v], 0.f));
      awB[r * 128 + v] = (__bf16)gx;
      awB[r * 128 + 64 + v] = (__bf16)mv;
    }
    __syncthreads();

    // ---- G3 (waves 0..3): xu = x_r@Wfr^T+b ; G4 (waves 4..7): beta ----
    if (w < 4) {
      int nt = w;
      v8f acc = {};
      acc = wmma_bf(load_afrag(arB, 64, lane, 0),  load_bfrag(Wfr, 0 * 4 + nt, lane), acc);
      acc = wmma_bf(load_afrag(arB, 64, lane, 32), load_bfrag(Wfr, 1 * 4 + nt, lane), acc);
      int n = nt * 16 + (lane & 15);
      int mrow0 = (lane & 16) ? 8 : 0;
      float bb = (n < V_N) ? b_fr[n] : 0.f;
#pragma unroll
      for (int i = 0; i < 8; ++i)
        xuS[(mrow0 + i) * 64 + n] = (n < V_N) ? (acc[i] + bb) : 0.f;
    } else {
      int nt = w - 4;
      v8f acc = {};
#pragma unroll
      for (int kt = 0; kt < 4; ++kt)
        acc = wmma_bf(load_afrag(awB, 128, lane, kt * 32), load_bfrag(Wwc, kt * 4 + nt, lane), acc);
      int n = nt * 16 + (lane & 15);
      int mrow0 = (lane & 16) ? 8 : 0;
      float bb = (n < V_N) ? b_wc[n] : 0.f;
#pragma unroll
      for (int i = 0; i < 8; ++i)
        btS[(mrow0 + i) * 64 + n] = (n < V_N) ? (acc[i] + bb) : 0.f;
    }
    __syncthreads();

    // ---- phase B: combine, loss, x_imp out, fill A5[x_imp|m] ----
    {
      float iv = invm[0];
      for (int i = tid; i < MTILE * V_N; i += THREADS) {
        int r = i / V_N, v = i - r * V_N;
        float bt = btS[r * 64 + v];
        float xc = bt * xuS[r * 64 + v] + (1.f - bt) * xhS[r * 64 + v];
        float xv = xg[i], mv = mg[i];
        lossAcc += fabsf(xv - xc) * mv * iv;
        float xi = mv * xv + (1.f - mv) * xc;
        out[gbase + (size_t)r * T_N * V_N + (size_t)t * V_N + v] = xi;
        A5[r * 640 + v]      = (__bf16)xi;
        A5[r * 640 + 64 + v] = (__bf16)mv;
      }
    }
    __syncthreads();

    // ---- G5: gates = [x_imp|m|h] @ [W_ih;W_hh]^T, fused LSTM update ----
    // A fragments are re-read from LDS per group (1 frag amortized over 4
    // WMMAs) to stay under 256 VGPRs -- no s_set_vgpr_msb churn.
    for (int grp = 0; grp < 4; ++grp) {
      int ht = w * 4 + grp;                         // hidden-col tile 0..31
      int nti = ht, ntf = ht + 32, ntg = ht + 64, nto = ht + 96;
      v8f ai = {}, afg = {}, agg = {}, aog = {};
#pragma unroll
      for (int kt = 0; kt < 20; ++kt) {
        v16bf a = load_afrag(A5, 640, lane, kt * 32);
        ai  = wmma_bf(a, load_bfrag(Wcat, kt * 128 + nti, lane), ai);
        afg = wmma_bf(a, load_bfrag(Wcat, kt * 128 + ntf, lane), afg);
        agg = wmma_bf(a, load_bfrag(Wcat, kt * 128 + ntg, lane), agg);
        aog = wmma_bf(a, load_bfrag(Wcat, kt * 128 + nto, lane), aog);
      }
      int hc = ht * 16 + (lane & 15);
      int mrow0 = (lane & 16) ? 8 : 0;
      float bi  = bsumS[hc];
      float bf_ = bsumS[512 + hc];
      float bg  = bsumS[1024 + hc];
      float bo  = bsumS[1536 + hc];
#pragma unroll
      for (int i = 0; i < 8; ++i) {
        int mr = mrow0 + i;
        float ig = sigm(ai[i] + bi);
        float fg = sigm(afg[i] + bf_);
        float gg = tanhf(agg[i] + bg);
        float og = sigm(aog[i] + bo);
        float cn = fg * cS[mr * H_N + hc] + ig * gg;
        cS[mr * H_N + hc] = cn;
        float hn = og * tanhf(cn);
        hS[mr * H_N + hc] = hn;
        outH[(size_t)(r0 + mr) * T_N * H_N + (size_t)t * H_N + hc] = hn;
      }
    }
    __syncthreads();
  }

  // ---- loss block-reduce + single global atomic ----
  red[tid] = lossAcc;
  __syncthreads();
  for (int s = THREADS / 2; s > 0; s >>= 1) {
    if (tid < s) red[tid] += red[tid + s];
    __syncthreads();
  }
  if (tid == 0) atomicAdd(out + XIMP_N, red[0]);
}

// ---------------- host launch ----------------
extern "C" void kernel_launch(void* const* d_in, const int* in_sizes, int n_in,
                              void* d_out, int out_size, void* d_ws, size_t ws_size,
                              hipStream_t stream) {
  (void)in_sizes; (void)n_in; (void)out_size; (void)ws_size;
  const float* x      = (const float*)d_in[0];
  const float* mask   = (const float*)d_in[1];
  const float* deltas = (const float*)d_in[2];
  const float* W_gh   = (const float*)d_in[3];
  const float* b_gh   = (const float*)d_in[4];
  const float* w_gx   = (const float*)d_in[5];
  const float* b_gx   = (const float*)d_in[6];
  const float* W_hist = (const float*)d_in[7];
  const float* b_hist = (const float*)d_in[8];
  const float* W_fr   = (const float*)d_in[9];
  const float* b_fr   = (const float*)d_in[10];
  const float* W_wc   = (const float*)d_in[11];
  const float* b_wc   = (const float*)d_in[12];
  const float* W_ih   = (const float*)d_in[13];
  const float* W_hh   = (const float*)d_in[14];
  const float* b_ih   = (const float*)d_in[15];
  const float* b_hh   = (const float*)d_in[16];
  float* out = (float*)d_out;

  char* ws = (char*)d_ws;
  float*  msum  = (float*)(ws + OFF_MSUM);
  __bf16* wgh   = (__bf16*)(ws + OFF_WGH);
  __bf16* whist = (__bf16*)(ws + OFF_WHIST);
  __bf16* wfr   = (__bf16*)(ws + OFF_WFR);
  __bf16* wwc   = (__bf16*)(ws + OFF_WWC);
  __bf16* wcat  = (__bf16*)(ws + OFF_WCAT);

  rits_init_kernel<<<1, 64, 0, stream>>>(msum, out);
  rits_msum_kernel<<<dim3(T_N, B_N / THREADS), THREADS, 0, stream>>>(mask, msum);

  // pack weight panels to bf16 fragment-major
  rits_pack_kernel<<<(2 * 32 * 512 + THREADS - 1) / THREADS, THREADS, 0, stream>>>(
      wgh, W_gh, nullptr, 0, 2, 32);
  rits_pack_kernel<<<(16 * 4 * 512 + THREADS - 1) / THREADS, THREADS, 0, stream>>>(
      whist, W_hist, nullptr, 1, 16, 4);
  rits_pack_kernel<<<(2 * 4 * 512 + THREADS - 1) / THREADS, THREADS, 0, stream>>>(
      wfr, W_fr, nullptr, 2, 2, 4);
  rits_pack_kernel<<<(4 * 4 * 512 + THREADS - 1) / THREADS, THREADS, 0, stream>>>(
      wwc, W_wc, nullptr, 3, 4, 4);
  rits_pack_kernel<<<(20 * 128 * 512 + THREADS - 1) / THREADS, THREADS, 0, stream>>>(
      wcat, W_ih, W_hh, 4, 20, 128);

  // persistent recurrent kernel: 128 blocks (16 rows each), 8 waves
  rits_main_kernel<<<B_N / MTILE, THREADS, SMEM_BYTES, stream>>>(
      x, mask, deltas, b_gh, w_gx, b_gx, b_hist, b_fr, b_wc, b_ih, b_hh,
      wgh, whist, wfr, wwc, wcat, msum, out);
}
